// HGRNBitAttention_12386685682227
// MI455X (gfx1250) — 8-bit matrix operand order repaired and checked by reading, compiled, not run on hardware
//
#include <hip/hip_runtime.h>

typedef __attribute__((ext_vector_type(8))) int v8i;

static constexpr int kD      = 1024;
static constexpr int kTok    = 8192;
static constexpr float kEpsBit = 1e-8f;
static constexpr float kEpsOut = 1.1920929e-07f;

__device__ __forceinline__ float blk_reduce(float v, bool domax) {
    __shared__ float buf[8];
    const int t = threadIdx.x;
    #pragma unroll
    for (int o = 16; o; o >>= 1) {
        float u = __shfl_xor(v, o, 32);
        v = domax ? fmaxf(v, u) : (v + u);
    }
    if ((t & 31) == 0) buf[t >> 5] = v;
    __syncthreads();
    if (t == 0) {
        float r = buf[0];
        #pragma unroll
        for (int i = 1; i < 8; ++i) r = domax ? fmaxf(r, buf[i]) : (r + buf[i]);
        buf[0] = r;
    }
    __syncthreads();
    float r = buf[0];
    __syncthreads();
    return r;
}

__device__ __forceinline__ char q8(float v, float sc) {
    float r = rintf(v * sc);
    r = fminf(fmaxf(r, -128.f), 127.f);
    return (char)(int)r;
}

__global__ __launch_bounds__(256) void hgrn_wscale_kernel(
    const float* __restrict__ Wi, const float* __restrict__ Wf,
    const float* __restrict__ Wg, const float* __restrict__ Wo,
    float* __restrict__ scales)
{
    const float* W = (blockIdx.x == 0) ? Wi : (blockIdx.x == 1) ? Wf
                    : (blockIdx.x == 2) ? Wg : Wo;
    float s = 0.f;
    for (int i = threadIdx.x; i < kD * kD; i += 256) s += fabsf(W[i]);
    s = blk_reduce(s, false);
    if (threadIdx.x == 0)
        scales[blockIdx.x] = fmaxf(s / (float)(kD * kD), 1e-5f);
}

__global__ __launch_bounds__(256) void hgrn_wquant_kernel(
    const float* __restrict__ Wi, const float* __restrict__ Wf,
    const float* __restrict__ Wg, const float* __restrict__ Wo,
    const float* __restrict__ scales,
    char* __restrict__ Qi, char* __restrict__ Qf,
    char* __restrict__ Qg, char* __restrict__ Qo)
{
    const int w = blockIdx.y;
    const float* W = (w == 0) ? Wi : (w == 1) ? Wf : (w == 2) ? Wg : Wo;
    char*        Q = (w == 0) ? Qi : (w == 1) ? Qf : (w == 2) ? Qg : Qo;
    const float inv = 1.0f / scales[w];
    const int idx = (blockIdx.x * 256 + threadIdx.x) * 4;
    float4 v = *(const float4*)(W + idx);
    char4 q;
    q.x = (char)(int)fminf(fmaxf(rintf(v.x * inv), -1.f), 1.f);
    q.y = (char)(int)fminf(fmaxf(rintf(v.y * inv), -1.f), 1.f);
    q.z = (char)(int)fminf(fmaxf(rintf(v.z * inv), -1.f), 1.f);
    q.w = (char)(int)fminf(fmaxf(rintf(v.w * inv), -1.f), 1.f);
    *(char4*)(Q + idx) = q;
}

__global__ __launch_bounds__(256) void hgrn_actquant_kernel(
    const float* __restrict__ x,
    const float* __restrict__ ni, const float* __restrict__ nf,
    const float* __restrict__ ng,
    char* __restrict__ qi, char* __restrict__ qf, char* __restrict__ qg,
    float* __restrict__ ri, float* __restrict__ rf, float* __restrict__ rg)
{
    const int tok = blockIdx.x;
    const int tid = threadIdx.x;
    const float4 xv = ((const float4*)(x + (size_t)tok * kD))[tid];
    float ss = xv.x * xv.x + xv.y * xv.y + xv.z * xv.z + xv.w * xv.w;
    ss = blk_reduce(ss, false);
    const float rs = rsqrtf(ss / (float)kD + kEpsBit);

    const float* nws[3] = { ni, nf, ng };
    char*  qo[3] = { qi, qf, qg };
    float* ro[3] = { ri, rf, rg };

    #pragma unroll
    for (int b = 0; b < 3; ++b) {
        const float4 nv = ((const float4*)nws[b])[tid];
        float4 xn;
        xn.x = xv.x * rs * nv.x;  xn.y = xv.y * rs * nv.y;
        xn.z = xv.z * rs * nv.z;  xn.w = xv.w * rs * nv.w;
        float am = fmaxf(fmaxf(fabsf(xn.x), fabsf(xn.y)),
                         fmaxf(fabsf(xn.z), fabsf(xn.w)));
        am = blk_reduce(am, true);
        const float amc = fmaxf(am, 1e-5f);
        const float sc  = 127.f / amc;
        char4 q;
        q.x = q8(xn.x, sc); q.y = q8(xn.y, sc);
        q.z = q8(xn.z, sc); q.w = q8(xn.w, sc);
        ((char4*)(qo[b] + (size_t)tok * kD))[tid] = q;
        if (tid == 0) ro[b][tok] = amc / 127.f;
    }
}

static constexpr int BM = 128, BN = 128, BK = 64;

__global__ __launch_bounds__(256) void hgrn_gemm_i8_kernel(
    const char*  __restrict__ Aq,
    const char*  __restrict__ Wq,
    const float* __restrict__ arecip,
    const float* __restrict__ wscale_ptr,
    float* __restrict__ Y)
{
    __shared__ char As[2][BM * BK];
    __shared__ char Bs[2][BN * BK];

    const int m0   = blockIdx.x * BM;
    const int n0   = blockIdx.y * BN;
    const int tid  = threadIdx.x;
    const int lane = tid & 31;
    const int wave = tid >> 5;
    const int wm   = wave & 1;
    const int wn   = wave >> 1;
    const int half = lane >> 4;
    const int lrow = lane & 15;

    v8i acc[4][2];
    #pragma unroll
    for (int mi = 0; mi < 4; ++mi)
        #pragma unroll
        for (int ni = 0; ni < 2; ++ni)
            #pragma unroll
            for (int r = 0; r < 8; ++r) acc[mi][ni][r] = 0;

    const int ldrow = tid >> 1;
    const int ldcol = (tid & 1) * 32;

    auto issue_slab = [&](int kk, int b) {
        const char* ga = Aq + (size_t)(m0 + ldrow) * kD + kk + ldcol;
        const char* gb = Wq + (size_t)(n0 + ldrow) * kD + kk + ldcol;
        unsigned la = (unsigned)(size_t)&As[b][ldrow * BK + ldcol];
        unsigned lb = (unsigned)(size_t)&Bs[b][ldrow * BK + ldcol];
        asm volatile("global_load_async_to_lds_b128 %0, %1, off"
                     :: "v"(la), "v"(ga) : "memory");
        asm volatile("global_load_async_to_lds_b128 %0, %1, off offset:16"
                     :: "v"(la), "v"(ga) : "memory");
        asm volatile("global_load_async_to_lds_b128 %0, %1, off"
                     :: "v"(lb), "v"(gb) : "memory");
        asm volatile("global_load_async_to_lds_b128 %0, %1, off offset:16"
                     :: "v"(lb), "v"(gb) : "memory");
    };

    int buf = 0;
    issue_slab(0, 0);

    for (int k0 = 0; k0 < kD; k0 += BK) {
        if (k0 + BK < kD) {
            issue_slab(k0 + BK, buf ^ 1);
            asm volatile("s_wait_asynccnt 0x4" ::: "memory");
        } else {
            asm volatile("s_wait_asynccnt 0x0" ::: "memory");
        }
        __syncthreads();

        #pragma unroll
        for (int mi = 0; mi < 4; ++mi) {
            union { v8i v; unsigned long long q[4]; } af;
            const char* arow = &As[buf][(wm * 64 + mi * 16 + lrow) * BK];
            const int kb = half * 16;
            af.q[0] = *(const unsigned long long*)(arow + kb);
            af.q[1] = *(const unsigned long long*)(arow + 8 + kb);
            af.q[2] = *(const unsigned long long*)(arow + 32 + kb);
            af.q[3] = *(const unsigned long long*)(arow + 40 + kb);
            #pragma unroll
            for (int ni = 0; ni < 2; ++ni) {
                union { v8i v; int4 h[2]; } bf;
                const char* brow = &Bs[buf][(wn * 32 + ni * 16 + lrow) * BK];
                bf.h[0] = *(const int4*)(brow + 16 * half);
                bf.h[1] = *(const int4*)(brow + 32 + 16 * half);
                acc[mi][ni] = __builtin_amdgcn_wmma_i32_16x16x64_iu8(
                     true, af.v,  true, bf.v,
                    acc[mi][ni],  false,  false);
            }
        }
        __syncthreads();
        buf ^= 1;
    }

    const float wdq = *wscale_ptr;
    #pragma unroll
    for (int mi = 0; mi < 4; ++mi)
        #pragma unroll
        for (int ni = 0; ni < 2; ++ni) {
            const int colg = n0 + wn * 32 + ni * 16 + lrow;
            #pragma unroll
            for (int r = 0; r < 8; ++r) {
                const int rowg = m0 + wm * 64 + mi * 16 + half * 8 + r;
                Y[(size_t)rowg * kD + colg] =
                    (float)acc[mi][ni][r] * wdq * arecip[rowg];
            }
        }
}

__global__ __launch_bounds__(256) void hgrn_gate_kernel(
    const float* __restrict__ yi, const float* __restrict__ yf,
    const float* __restrict__ yg,
    const float* __restrict__ nout, const float* __restrict__ no,
    char* __restrict__ oq, float* __restrict__ orecip)
{
    const int tok = blockIdx.x;
    const int tid = threadIdx.x;
    const float4 vi = ((const float4*)(yi + (size_t)tok * kD))[tid];
    const float4 vf = ((const float4*)(yf + (size_t)tok * kD))[tid];
    const float4 vg = ((const float4*)(yg + (size_t)tok * kD))[tid];

    float4 f;
    f.x = 1.f / (1.f + __expf(-vf.x)); f.y = 1.f / (1.f + __expf(-vf.y));
    f.z = 1.f / (1.f + __expf(-vf.z)); f.w = 1.f / (1.f + __expf(-vf.w));
    float4 p = f;
    #pragma unroll
    for (int k = 0; k < 11; ++k) {
        p.x *= p.x; p.y *= p.y; p.z *= p.z; p.w *= p.w;
    }
    float4 c;
    c.x = 1.f - p.x; c.y = 1.f - p.y; c.z = 1.f - p.z; c.w = 1.f - p.w;

    float s = c.x * vg.x + c.y * vg.y + c.z * vg.z + c.w * vg.w;
    s = blk_reduce(s, false);

    float4 o;
    o.x = vi.x * s; o.y = vi.y * s; o.z = vi.z * s; o.w = vi.w * s;

    float v1 = blk_reduce(o.x*o.x + o.y*o.y + o.z*o.z + o.w*o.w, false);
    const float rs1 = rsqrtf(v1 / (float)kD + kEpsOut);
    const float4 nn = ((const float4*)nout)[tid];
    float4 o2;
    o2.x = o.x * rs1 * nn.x; o2.y = o.y * rs1 * nn.y;
    o2.z = o.z * rs1 * nn.z; o2.w = o.w * rs1 * nn.w;

    float v2 = blk_reduce(o2.x*o2.x + o2.y*o2.y + o2.z*o2.z + o2.w*o2.w, false);
    const float rs2 = rsqrtf(v2 / (float)kD + kEpsBit);
    const float4 nb = ((const float4*)no)[tid];
    float4 o3;
    o3.x = o2.x * rs2 * nb.x; o3.y = o2.y * rs2 * nb.y;
    o3.z = o2.z * rs2 * nb.z; o3.w = o2.w * rs2 * nb.w;

    float am = fmaxf(fmaxf(fabsf(o3.x), fabsf(o3.y)),
                     fmaxf(fabsf(o3.z), fabsf(o3.w)));
    am = blk_reduce(am, true);
    const float amc = fmaxf(am, 1e-5f);
    const float sc  = 127.f / amc;
    char4 q;
    q.x = q8(o3.x, sc); q.y = q8(o3.y, sc);
    q.z = q8(o3.z, sc); q.w = q8(o3.w, sc);
    ((char4*)(oq + (size_t)tok * kD))[tid] = q;
    if (tid == 0) orecip[tok] = amc / 127.f;
}

extern "C" void kernel_launch(void* const* d_in, const int* in_sizes, int n_in,
                              void* d_out, int out_size, void* d_ws, size_t ws_size,
                              hipStream_t stream)
{
    (void)in_sizes; (void)n_in; (void)out_size; (void)ws_size;
    const float* x    = (const float*)d_in[0];
    const float* Wi   = (const float*)d_in[1];
    const float* Wf   = (const float*)d_in[2];
    const float* Wg   = (const float*)d_in[3];
    const float* Wo   = (const float*)d_in[4];
    const float* ni   = (const float*)d_in[5];
    const float* nf   = (const float*)d_in[6];
    const float* ng   = (const float*)d_in[7];
    const float* no   = (const float*)d_in[8];
    const float* nout = (const float*)d_in[9];

    char* ws = (char*)d_ws;
    const size_t MB = 1u << 20;
    char*  xq_i  = ws;
    char*  xq_f  = ws + 8  * MB;
    char*  xq_g  = ws + 16 * MB;
    char*  wq_i  = ws + 24 * MB;
    char*  wq_f  = ws + 25 * MB;
    char*  wq_g  = ws + 26 * MB;
    char*  wq_o  = ws + 27 * MB;
    char*  oq    = ws + 28 * MB;
    float* y_i   = (float*)(ws + 36  * MB);
    float* y_f   = (float*)(ws + 68  * MB);
    float* y_g   = (float*)(ws + 100 * MB);
    float* wsc   = (float*)(ws + 132 * MB);
    float* r_i   = (float*)(ws + 132 * MB + 1024);
    float* r_f   = r_i + kTok;
    float* r_g   = r_f + kTok;
    float* r_o   = r_g + kTok;

    hgrn_wscale_kernel<<<4, 256, 0, stream>>>(Wi, Wf, Wg, Wo, wsc);
    hgrn_wquant_kernel<<<dim3(kD * kD / 1024, 4), 256, 0, stream>>>(
        Wi, Wf, Wg, Wo, wsc, wq_i, wq_f, wq_g, wq_o);

    hgrn_actquant_kernel<<<kTok, 256, 0, stream>>>(
        x, ni, nf, ng, xq_i, xq_f, xq_g, r_i, r_f, r_g);

    dim3 gg(kTok / BM, kD / BN);
    hgrn_gemm_i8_kernel<<<gg, 256, 0, stream>>>(xq_i, wq_i, r_i, wsc + 0, y_i);
    hgrn_gemm_i8_kernel<<<gg, 256, 0, stream>>>(xq_f, wq_f, r_f, wsc + 1, y_f);
    hgrn_gemm_i8_kernel<<<gg, 256, 0, stream>>>(xq_g, wq_g, r_g, wsc + 2, y_g);

    hgrn_gate_kernel<<<kTok, 256, 0, stream>>>(y_i, y_f, y_g, nout, no, oq, r_o);

    hgrn_gemm_i8_kernel<<<gg, 256, 0, stream>>>(oq, wq_o, r_o, wsc + 3,
                                                (float*)d_out);
}
